// Loss_33157147525547
// MI455X (gfx1250) — matrix operand order repaired and checked by reading, not compiled, not run on hardware
//
#include <hip/hip_runtime.h>

// Problem constants (B,T,C,H,W) = (4,24,4,192,192)
#define B_    4
#define T_    24
#define C_    4
#define H_    192
#define W_    192
#define HW_   (H_*W_)        // 36864
#define CHW_  (C_*HW_)       // 147456
#define NF_   (B_*T_)        // 96 frames
#define NPAIR_ (B_*(T_-1))   // 92 pairs
#define PAD_  19             // int(0.1*192)
#define S_    39             // 2*PAD_+1 offsets
#define MT_   48             // 3x16 WMMA tiles covering 39
#define HP_   240            // padded rows:  h in [-19, 219] -> hp = h+19
#define WP_   256            // padded cols:  w in [-28, 210] -> wp = w+32 (16B-aligned A loads)
#define HOFF_ 19
#define WOFF_ 32
#define FCH_  (HP_*WP_)      // 61440 elems per channel
#define FST_  (C_*FCH_)      // 245760 elems per frame

#define HB_   48             // h-rows per workgroup
#define NHB_  (H_/HB_)       // 4 h-blocks
#define NSUB_ (C_*NHB_)      // 16 partials per pair
#define AROWS_ (HB_+MT_-1)   // 95 a-rows staged per WG
#define AW_   192            // staged a-row width (wp-32 in [0,192))

typedef __attribute__((ext_vector_type(8)))  float          v8f;
typedef __attribute__((ext_vector_type(16))) __bf16         v16bf;
typedef __attribute__((ext_vector_type(16))) unsigned short v16u;
typedef __attribute__((ext_vector_type(8)))  unsigned short v8u;
typedef __attribute__((ext_vector_type(4)))  unsigned int   v4u32;
typedef __attribute__((ext_vector_type(8)))  int            v8i32;
typedef __attribute__((ext_vector_type(4)))  int            v4i32;

#if defined(__has_builtin)
#  if __has_builtin(__builtin_amdgcn_tensor_load_to_lds) && __has_builtin(__builtin_amdgcn_s_wait_tensorcnt)
#    define USE_TDM 1
#  endif
#endif

__device__ __forceinline__ unsigned short f2bf_rne(float f) {
    unsigned int u = __builtin_bit_cast(unsigned int, f);
    u += 0x7FFFu + ((u >> 16) & 1u);            // round-to-nearest-even
    return (unsigned short)(u >> 16);
}

#ifdef USE_TDM
// Issue one TDM 2D tile load: rows x cols (bf16 elems), row stride in elems.
// D# packing per CDNA5 ISA ch.8: group0 = {flags, lds_addr, global_addr, type},
// group1 = {data_size, tensor dims, tile dims, dim0 stride}. Groups 2/3/4 zero (2D).
__device__ __forceinline__ void tdm_load_2d(const unsigned short* gsrc, unsigned lds_byte_off,
                                            unsigned rows, unsigned cols, unsigned strideElems) {
    unsigned long long ga = (unsigned long long)(uintptr_t)gsrc;
    v4u32 g0;
    g0[0] = 1u;                                     // count=1, user mode
    g0[1] = lds_byte_off;                           // lds_addr [63:32]
    g0[2] = (unsigned)(ga & 0xFFFFFFFFu);           // global_addr lo
    g0[3] = (unsigned)((ga >> 32) & 0x01FFFFFFu) | 0x80000000u;  // addr hi | type=2
    v8i32 g1;
    g1[0] = (int)(1u << 16);                        // data_size=1 (2 bytes)
    g1[1] = (int)((cols & 0xFFFFu) << 16);          // tensor_dim0[15:0] (bits 79:48 base)
    g1[2] = (int)((cols >> 16) | ((rows & 0xFFFFu) << 16));      // tdim0 hi | tdim1 lo
    g1[3] = (int)((rows >> 16) | (cols << 16));     // tdim1 hi | tile_dim0
    g1[4] = (int)(rows & 0xFFFFu);                  // tile_dim1 | tile_dim2=0
    g1[5] = (int)strideElems;                       // tensor_dim0_stride lo32
    g1[6] = 0;                                      // stride0 hi16 | stride1 lo16
    g1[7] = 0;                                      // stride1 hi
    v4i32 gz4 = {};
    v8i32 gz8 = {};
    __builtin_amdgcn_tensor_load_to_lds(g0, g1, gz4, gz4, gz8, 0);
}
#endif

// ---------------- per-frame mean / inv-std ----------------
__global__ __launch_bounds__(256) void stats_kernel(const float* __restrict__ renders,
                                                    float* __restrict__ stats) {
    __shared__ float sh[256];
    const int f = blockIdx.x;
    const size_t base = (size_t)f * CHW_;
    float s = 0.f, s2 = 0.f;
    for (int i = threadIdx.x; i < CHW_; i += 256) {
        float x = renders[base + i];
        s += x; s2 += x * x;
    }
    sh[threadIdx.x] = s; __syncthreads();
    for (int st = 128; st > 0; st >>= 1) { if (threadIdx.x < st) sh[threadIdx.x] += sh[threadIdx.x + st]; __syncthreads(); }
    float S = sh[0]; __syncthreads();
    sh[threadIdx.x] = s2; __syncthreads();
    for (int st = 128; st > 0; st >>= 1) { if (threadIdx.x < st) sh[threadIdx.x] += sh[threadIdx.x + st]; __syncthreads(); }
    float S2 = sh[0];
    if (threadIdx.x == 0) {
        float mean = S / (float)CHW_;
        float var  = S2 / (float)CHW_ - mean * mean;
        var = var > 0.f ? var : 0.f;
        float sd = sqrtf(var);
        if (sd == 0.f) sd = 1.f;                 // std += (std==0)
        stats[2 * f]     = mean;
        stats[2 * f + 1] = 1.f / sd;
    }
}

// -------- normalized bf16 frames into zero-padded buffer --------
__global__ __launch_bounds__(256) void pad_kernel(const float* __restrict__ renders,
                                                  const float* __restrict__ stats,
                                                  unsigned short* __restrict__ norm) {
    const int blk = blockIdx.x;                  // f*(C_*HP_) + c*HP_ + hp
    const int hp  = blk % HP_;
    const int c   = (blk / HP_) % C_;
    const int f   = blk / (HP_ * C_);
    const int wp  = threadIdx.x;                 // 0..255 == WP_
    const int h = hp - HOFF_;
    const int w = wp - WOFF_;
    unsigned short v = 0;
    if ((unsigned)h < (unsigned)H_ && (unsigned)w < (unsigned)W_) {
        float mean = stats[2 * f], inv = stats[2 * f + 1];
        float x = renders[(((size_t)f * C_ + c) * H_ + h) * (size_t)W_ + w];
        v = f2bf_rne((x - mean) * inv);
    }
    norm[((size_t)f * C_ + c) * FCH_ + (size_t)hp * WP_ + wp] = v;
}

// ---------------- supervised masked-L1 partials ----------------
__global__ __launch_bounds__(256) void sup_kernel(const float* __restrict__ frames,
                                                  const float* __restrict__ renders,
                                                  float* __restrict__ sup) {
    __shared__ float sh[256];
    const int bt = blockIdx.x;
    const size_t base = (size_t)bt * CHW_;
    float s_in = 0.f, s_out = 0.f, s_rgb = 0.f, cnt = 0.f;
    for (int i = threadIdx.x; i < HW_; i += 256) {
        float ga = frames [base + 3 * (size_t)HW_ + i];
        float ra = renders[base + 3 * (size_t)HW_ + i];
        float m = ga > 0.f ? 1.f : 0.f;
        float d = fabsf(ga - ra);
        s_in  += d * m;
        s_out += d * (1.f - m);
        cnt   += m;
        #pragma unroll
        for (int ch = 0; ch < 3; ++ch) {
            float g = frames [base + (size_t)ch * HW_ + i];
            float r = renders[base + (size_t)ch * HW_ + i];
            s_rgb += fabsf(g * ga - r * ra) * m;
        }
    }
    float vals[4] = { s_in, s_out, s_rgb, cnt };
    float red[4];
    for (int k = 0; k < 4; ++k) {
        sh[threadIdx.x] = vals[k]; __syncthreads();
        for (int st = 128; st > 0; st >>= 1) { if (threadIdx.x < st) sh[threadIdx.x] += sh[threadIdx.x + st]; __syncthreads(); }
        red[k] = sh[0]; __syncthreads();
    }
    if (threadIdx.x == 0) {
        float cin = red[3], cout = (float)HW_ - red[3];
        sup[bt] = red[0] / cin + red[1] / cout + red[2] / cin;   // alpha_in + alpha_out + rgb
    }
}

// ---------------- WMMA correlation: cc[sy,sx] = sum a[c,h+sy,w']*b[c,h,w'-sx] ----------------
// grid: NPAIR_*16 blocks (16 = 4 channels x 4 h-blocks); block: 288 threads = 9 waves,
// wave (mi,ni) owns the 16x16 tile at (sy = mi*16.., sx = ni*16..).
// LDS-staged via Tensor Data Mover (TDM) when available: each WG's unique ~60KB
// footprint is DMA'd to LDS once (2 tensor_load_to_lds + s_wait_tensorcnt),
// then all sliding-window fragment traffic is served from the 320KB WGP LDS.
__global__ __launch_bounds__(288) void corr_kernel(const unsigned short* __restrict__ norm,
                                                   float* __restrict__ corr) {
    __shared__ __align__(16) unsigned short sA[AROWS_ * AW_];  // 95*192*2 = 36,480 B
    __shared__ __align__(16) unsigned short sB[HB_ * WP_];     // 48*256*2 = 24,576 B

    const int pair = blockIdx.x >> 4;
    const int sub  = blockIdx.x & 15;
    const int c    = sub >> 2;
    const int hb   = sub & 3;
    const int bb = pair / (T_ - 1);
    const int tt = pair % (T_ - 1);
    const int fa = bb * T_ + tt;          // 'a' = renders[:, :-1]
    const int fb = fa + 1;                // 'b' = renders[:, 1:]
    const unsigned short* Abase = norm + (size_t)fa * FST_ + (size_t)c * FCH_;
    const unsigned short* Bbase = norm + (size_t)fb * FST_ + (size_t)c * FCH_;
    const int h0 = hb * HB_;

#ifdef USE_TDM
    if (threadIdx.x < 32) {               // single wave issues both DMA descriptors
        tdm_load_2d(Abase + (size_t)h0 * WP_ + WOFF_,
                    (unsigned)(uintptr_t)(&sA[0]) & 0xFFFFFFFFu,
                    AROWS_, AW_, WP_);
        tdm_load_2d(Bbase + (size_t)h0 * WP_,
                    (unsigned)(uintptr_t)(&sB[0]) & 0xFFFFFFFFu,
                    HB_, WP_, WP_);
        __builtin_amdgcn_s_wait_tensorcnt(0);
    }
#else
    {   // fallback: cooperative staging (v8u = 16B chunks)
        v8u*       dA = (v8u*)sA;
        const int  nA = AROWS_ * (AW_ / 8);                    // 95*24 = 2280 chunks
        for (int i = threadIdx.x; i < nA; i += 288) {
            int row = i / (AW_ / 8), c8 = i % (AW_ / 8);
            dA[i] = *(const v8u*)(Abase + (size_t)(h0 + row) * WP_ + WOFF_ + c8 * 8);
        }
        v8u*       dB = (v8u*)sB;
        const int  nB = HB_ * (WP_ / 8);                       // 48*32 = 1536 chunks
        for (int i = threadIdx.x; i < nB; i += 288) {
            int row = i / (WP_ / 8), c8 = i % (WP_ / 8);
            dB[i] = *(const v8u*)(Bbase + (size_t)(h0 + row) * WP_ + c8 * 8);
        }
    }
#endif
    __syncthreads();

    const int wave = threadIdx.x >> 5;
    const int lane = threadIdx.x & 31;
    const int l16  = lane & 15;
    const int hi   = lane >> 4;           // 0 / 1
    const int mi = wave % 3;              // sy tile
    const int ni = wave / 3;              // sx tile

    v8f acc = {};
    const int aRow = mi * 16 + l16;                       // sA row = hl + aRow
    // B lane base col: wp = 32j + K + (WOFF_+PAD_) - ni*16 - n, K = hi*8 + e (+16 for e>=8)
    const int bCol = (WOFF_ + PAD_) + hi * 8 - ni * 16 - l16;   // in [4, 59]

    for (int hl = 0; hl < HB_; ++hl) {
        const unsigned short* ap0 = sA + (hl + aRow) * AW_ + hi * 8;   // wp-32 = 32j + kk
        const unsigned short* bp0 = sB + hl * WP_ + bCol;
        #pragma unroll
        for (int j = 0; j < 6; ++j) {
            // A fragment 16x32 bf16: lanes<16 hold K {0..7,16..23}, lanes>=16 {8..15,24..31}
            const unsigned short* ap = ap0 + 32 * j;
            v8u alo = *(const v8u*)(ap);          // 16B aligned by construction
            v8u ahi = *(const v8u*)(ap + 16);
            v16u au;
            #pragma unroll
            for (int e = 0; e < 8; ++e) { au[e] = alo[e]; au[8 + e] = ahi[e]; }
            // B fragment 32x16 bf16: lanes<16 hold K {0..7,16..23}, lanes>=16 {8..15,24..31} (sliding)
            const unsigned short* bp = bp0 + 32 * j;
            v16u bu;
            #pragma unroll
            for (int e = 0; e < 8; ++e) { bu[e] = bp[e]; bu[8 + e] = bp[16 + e]; }

            v16bf av = __builtin_bit_cast(v16bf, au);
            v16bf bv = __builtin_bit_cast(v16bf, bu);
            acc = __builtin_amdgcn_wmma_f32_16x16x32_bf16(false, av, false, bv,
                                                          (short)0, acc, false, false);
        }
    }
    // D layout: VGPR r -> M = r (+8 for hi half), N = l16
    float* out = corr + (size_t)blockIdx.x * (MT_ * MT_);
    #pragma unroll
    for (int r = 0; r < 8; ++r) {
        int sy = mi * 16 + r + hi * 8;
        int sx = ni * 16 + l16;
        out[sy * MT_ + sx] = acc[r];
    }
}

// ---------------- final combine: max/mean + supervised ----------------
__global__ __launch_bounds__(256) void fin_kernel(const float* __restrict__ corr,
                                                  const float* __restrict__ sup,
                                                  float* __restrict__ out) {
    __shared__ float sh[256];
    const int tid = threadIdx.x;
    float accum = 0.f;                                   // meaningful on tid 0
    for (int pair = 0; pair < NPAIR_; ++pair) {
        const float* p = corr + (size_t)pair * NSUB_ * (MT_ * MT_);
        float lmax = -3.4e38f;
        for (int idx = tid; idx < S_ * S_; idx += 256) { // only the valid 39x39 region
            int sy = idx / S_, sx = idx % S_;
            float s = 0.f;
            #pragma unroll
            for (int k = 0; k < NSUB_; ++k) s += p[k * (MT_ * MT_) + sy * MT_ + sx];
            lmax = fmaxf(lmax, s);
        }
        sh[tid] = lmax; __syncthreads();
        for (int st = 128; st > 0; st >>= 1) { if (tid < st) sh[tid] = fmaxf(sh[tid], sh[tid + st]); __syncthreads(); }
        if (tid == 0) accum += sh[0];
        __syncthreads();
    }
    float sv = (tid < NF_) ? sup[tid] : 0.f;
    sh[tid] = sv; __syncthreads();
    for (int st = 128; st > 0; st >>= 1) { if (tid < st) sh[tid] += sh[tid + st]; __syncthreads(); }
    if (tid == 0) {
        float supervised = sh[0] / (float)NF_;
        float zmean = (accum / (float)NPAIR_) / (float)CHW_;
        out[0] = supervised + 10.f * 0.5f * (1.f - zmean);   // TC_WEIGHT * (1-zncc)/2
    }
}

extern "C" void kernel_launch(void* const* d_in, const int* in_sizes, int n_in,
                              void* d_out, int out_size, void* d_ws, size_t ws_size,
                              hipStream_t stream) {
    const float* frames  = (const float*)d_in[0];
    const float* renders = (const float*)d_in[1];
    float* out = (float*)d_out;

    char* ws = (char*)d_ws;
    unsigned short* norm = (unsigned short*)ws;                   // 96*245760*2 B = 47,185,920 B
    size_t normBytes = (size_t)NF_ * FST_ * 2;
    float* corr  = (float*)(ws + normBytes);                      // 1472*2304*4 B = 13,565,952 B
    size_t corrBytes = (size_t)NPAIR_ * NSUB_ * (MT_ * MT_) * 4;
    float* stats = (float*)(ws + normBytes + corrBytes);          // 192 floats
    float* sup   = stats + 2 * NF_;                               // 96 floats

    stats_kernel<<<NF_, 256, 0, stream>>>(renders, stats);
    pad_kernel<<<NF_ * C_ * HP_, 256, 0, stream>>>(renders, stats, norm);
    sup_kernel<<<NF_, 256, 0, stream>>>(frames, renders, sup);
    corr_kernel<<<NPAIR_ * NSUB_, 288, 0, stream>>>(norm, corr);
    fin_kernel<<<1, 256, 0, stream>>>(corr, sup, out);
}
